// ProbAttention_7550552506918
// MI455X (gfx1250) — matrix operand order repaired and checked by reading, not compiled, not run on hardware
//
#include <hip/hip_runtime.h>

// ProbSparse attention sampling stage (Informer), MI455X / gfx1250.
// Outputs (concatenated): v = transpose(values) -> (B,H,L_K,D)  [4,194,304 f32]
//                         M = max_u(QK) - sum_u(QK)/L_K -> (B,H,L_Q) [65,536 f32]

typedef __attribute__((ext_vector_type(16))) _Float16 v16h;
typedef __attribute__((ext_vector_type(8)))  float    v8f;

#define B_  2
#define H_  8
#define LQ_ 4096
#define LK_ 4096
#define D_  64
#define U_  45

// ---------------------------------------------------------------------------
// Kernel 1: values (B, L_K, H, D) -> v (B, H, L_K, D).  Pure float4 permute
// copy; read side is perfectly linear/coalesced, write side coalesced in
// 256-byte rows. 33 MB total traffic -> ~1.4 us at 23.3 TB/s.
// ---------------------------------------------------------------------------
__global__ __launch_bounds__(256) void
probattn_transpose_v(const float* __restrict__ values, float* __restrict__ vout) {
  int tid = blockIdx.x * 256 + threadIdx.x;       // one float4 per thread
  int d4 = tid & 15;                              // float4 index within 64-f row
  int t  = tid >> 4;                              // row id in (b, l, h) order
  int h  = t & (H_ - 1);
  t >>= 3;
  int l  = t & (LK_ - 1);
  int b  = t >> 12;
  const float4* src = (const float4*)values;
  float4*       dst = (float4*)vout;
  float4 val = src[tid];                          // linear read
  dst[(((b * H_ + h) * LK_ + l) << 4) + d4] = val;
}

// ---------------------------------------------------------------------------
// Kernel 2: M[b,h,q].  One wave per query.  The query row is replicated across
// all 16 rows of the WMMA A matrix; 16 gathered keys form B columns, so every
// row of D = A x B holds q . key_n and acc[0] in lane L is the dot for key
// n = L & 15.  Two chained v_wmma_f32_16x16x32_f16 cover D=64; three column
// tiles cover U=45 (invalid columns zeroed -> sum unaffected; max lane-masked).
// EXEC is all-ones at every WMMA (masking is data-level only).
// ---------------------------------------------------------------------------
__global__ __launch_bounds__(256) void
probattn_sample_scores(const float* __restrict__ Q,
                       const float* __restrict__ K,
                       const int*   __restrict__ idxs,   // (L_Q, U) int32
                       float*       __restrict__ Mout) { // (B, H, L_Q)
  const int lane = threadIdx.x & 31;
  const int wave = threadIdx.x >> 5;
  const int gq   = blockIdx.x * 8 + wave;         // (b*H + h)*L_Q + q
  const int q    = gq & (LQ_ - 1);
  const int bh   = gq >> 12;
  const int h    = bh & (H_ - 1);
  const int b    = bh >> 3;

  const int n  = lane & 15;                       // B column / key slot
  const int hi = lane >> 4;                       // lane-half selector

  // ---- A fragments: query replicated across all 16 rows -------------------
  // A layout (16-bit A 16x32): elem i -> K = i + hi*16 (matches B).
  const float* qrow = Q + (((size_t)(b * LQ_ + q) * H_ + h) << 6);
  v16h a0, a1;
#pragma unroll
  for (int i = 0; i < 16; ++i) {
    int dd = hi * 16 + i;
    a0[i] = (_Float16)qrow[dd];                   // d in [0,32)
    a1[i] = (_Float16)qrow[dd + 32];              // d in [32,64)
  }

  float totMax = -__builtin_inff();
  float totSum = 0.0f;

  for (int t = 0; t < 3; ++t) {                   // 3 tiles of 16 keys
    const int  u     = t * 16 + n;
    const bool valid = (u < U_);

    // ---- B fragments: contiguous k order, same as A (16-bit B 32x16): elem i -> K = i + hi*16.
    // Contiguous 16-float run of the gathered key row -> b128 loads.
    v16h b0, b1;
    if (valid) {
      const int kk = idxs[q * U_ + u];            // gather index (shared b,h)
      const float* krow = K + (((size_t)(b * LK_ + kk) * H_ + h) << 6);
#pragma unroll
      for (int i = 0; i < 16; ++i) {
        b0[i] = (_Float16)krow[hi * 16 + i];      // d in [0,32)
        b1[i] = (_Float16)krow[32 + hi * 16 + i]; // d in [32,64)
      }
    } else {
#pragma unroll
      for (int i = 0; i < 16; ++i) {              // zero column: dot == 0
        b0[i] = (_Float16)0.0f;
        b1[i] = (_Float16)0.0f;
      }
    }

    // ---- D = A x B (+C), f32 accumulate over full D=64 ----------------------
    v8f acc = {};
    acc = __builtin_amdgcn_wmma_f32_16x16x32_f16(false, a0, false, b0,
                                                 (short)0, acc, false, false);
    acc = __builtin_amdgcn_wmma_f32_16x16x32_f16(false, a1, false, b1,
                                                 (short)0, acc, false, false);

    // acc[0]: lanes 0-15 hold (M=0, N=lane), lanes 16-31 (M=8, N=lane-16);
    // all rows identical, so this is q . key_n in every lane.
    const float val  = acc[0];
    float mval = valid ? val : -__builtin_inff();
    float sval = valid ? val : 0.0f;

    // Butterfly reduce within each 16-lane half (masks 1,2,4,8 stay in-half;
    // lanes 16-31 are duplicates, so summing across halves would double-count).
#pragma unroll
    for (int m = 1; m < 16; m <<= 1) {
      mval = fmaxf(mval, __shfl_xor(mval, m, 32));
      sval += __shfl_xor(sval, m, 32);
    }
    totMax = fmaxf(totMax, mval);
    totSum += sval;
  }

  if (lane == 0) {
    Mout[gq] = totMax - totSum * (1.0f / (float)LK_);
  }
}

// ---------------------------------------------------------------------------
extern "C" void kernel_launch(void* const* d_in, const int* in_sizes, int n_in,
                              void* d_out, int out_size, void* d_ws, size_t ws_size,
                              hipStream_t stream) {
  const float* queries = (const float*)d_in[0];   // (B, L_Q, H, D) f32
  const float* keys    = (const float*)d_in[1];   // (B, L_K, H, D) f32
  const float* values  = (const float*)d_in[2];   // (B, L_K, H, D) f32
  const int*   idxs    = (const int*)d_in[3];     // (L_Q, U) int32

  float* out   = (float*)d_out;
  float* v_out = out;                                      // B*H*L_K*D floats
  float* m_out = out + (size_t)B_ * H_ * LK_ * D_;         // B*H*L_Q floats

  // v transpose: B*L_K*H*16 float4 = 1,048,576 threads.
  probattn_transpose_v<<<4096, 256, 0, stream>>>(values, v_out);

  // M: one wave per query; B*H*L_Q = 65,536 waves, 8 waves per 256-thr block.
  probattn_sample_scores<<<8192, 256, 0, stream>>>(queries, keys, idxs, m_out);
}
